// LFADS_1975684956401
// MI455X (gfx1250) — matrix operand order repaired and checked by reading, not compiled, not run on hardware
//
#include <hip/hip_runtime.h>
#include <math.h>

// ---------------- problem constants ----------------
#define Bz  128
#define Tz  400
#define Nz  256
#define EHz 512
#define CHz 256
#define GHz 512
#define ICz 64
#define Uz  64
#define FDz 64

#define PANEL_LD 776          // max Ktot (768) + 8-half pad -> rows 16B aligned, banks spread
#define SA_LD    48           // 32 + 16-half pad
#define SA_TILE  (128 * SA_LD)

typedef __attribute__((ext_vector_type(16))) _Float16 v16h;
typedef __attribute__((ext_vector_type(8)))  float    v8f;

// ---------------- CDNA5 async global->LDS copy (ASYNCcnt tracked) ----------------
__device__ __forceinline__ void async_copy_b128(void* lds_ptr, const void* gptr) {
  unsigned loff = (unsigned)(uintptr_t)lds_ptr;   // low 32 bits of flat shared ptr = LDS byte offset
  asm volatile("global_load_async_to_lds_b128 %0, %1, off"
               :: "v"(loff), "v"(gptr) : "memory");
}
__device__ __forceinline__ void async_wait0() {
  asm volatile("s_wait_asynccnt 0x0" ::: "memory");
}
__device__ __forceinline__ void async_wait2() {
  asm volatile("s_wait_asynccnt 0x2" ::: "memory");
}

// ---------------- WMMA fragment loaders (wave32, 16x16x32 f16) ----------------
// A-matrix 16x32: lanes 0-15 -> M=0..15 with K 0..7 & 16..23 ; lanes 16-31 -> K 8..15 & 24..31
__device__ __forceinline__ v16h load_frag_a(const _Float16* base, int ld) {
  int lane = threadIdx.x & 31;
  int m  = lane & 15;
  int kb = (lane & 16) >> 1;              // 0 or 8
  const _Float16* p = base + m * ld + kb;
  union { v16h v; float4 f[2]; } u;
  u.f[0] = *(const float4*)(p);
  u.f[1] = *(const float4*)(p + 16);
  return u.v;
}
// B-matrix 32x16 read from a column-major (n-row) tile: lanes 0-15 -> K 0..7 & 16..23 ; 16-31 -> K 8..15 & 24..31
__device__ __forceinline__ v16h load_frag_b(const _Float16* baseT, int ld) {
  int lane = threadIdx.x & 31;
  int n  = lane & 15;
  int kb = (lane & 16) >> 1;              // 0 or 8
  const _Float16* p = baseT + n * ld + kb;
  union { v16h v; float4 f[2]; } u;
  u.f[0] = *(const float4*)(p);
  u.f[1] = *(const float4*)(p + 16);
  return u.v;
}

// ---------------- grid-wide sync (persistent cooperative kernels) ----------------
__device__ __forceinline__ void grid_sync(unsigned* counter, unsigned nwg) {
  __syncthreads();
  if (threadIdx.x == 0) {
    __threadfence();
    unsigned ticket = atomicAdd(counter, 1u);
    unsigned target = (ticket / nwg + 1u) * nwg;   // monotonic counter: no reset race
    volatile unsigned* vc = (volatile unsigned*)counter;
    while (*vc < target) { __builtin_amdgcn_s_sleep(1); }
    __threadfence();
  }
  __syncthreads();
}

// ---------------- small prep kernels ----------------
__global__ void zero_u32(unsigned* p, int n) {
  int i = blockIdx.x * blockDim.x + threadIdx.x;
  if (i < n) p[i] = 0u;
}

// f32 (K,N) -> f16 transposed (N,K)
__global__ void f2h_t_kernel(const float* __restrict__ in, _Float16* __restrict__ out,
                             int K, int N) {
  int idx = blockIdx.x * blockDim.x + threadIdx.x;
  if (idx >= K * N) return;
  int k = idx / N, n = idx % N;
  out[(size_t)n * K + k] = (_Float16)in[idx];
}

// x (B,T,N) f32 -> x_h (T,B,N) f16 (time-major for the scans)
__global__ void pack_x_kernel(const float* __restrict__ x, _Float16* __restrict__ xh) {
  int idx = blockIdx.x * blockDim.x + threadIdx.x;
  const int total = Tz * Bz * Nz;
  if (idx >= total) return;
  int n  = idx % Nz;
  int tb = idx / Nz;
  int b  = tb % Bz;
  int t  = tb / Bz;
  xh[idx] = (_Float16)x[((size_t)b * Tz + t) * Nz + n];
}

// Column-norm-constrain Wh (3,H,H), pack TRANSPOSED [Wh;Wx]^T -> Wt (3H rows, Ktot=H+In cols) f16.
__global__ void constrain_pack_kernel(const float* __restrict__ Wh,
                                      const float* __restrict__ Wx,
                                      _Float16* __restrict__ Wt, int H, int In) {
  int tid = blockIdx.x * blockDim.x + threadIdx.x;
  if (tid >= 3 * H) return;
  int g = tid / H, j = tid % H;
  int Ktot = H + In;
  const float* Whg = Wh + (size_t)g * H * H;
  float ss = 0.f;
  for (int i = 0; i < H; ++i) { float v = Whg[(size_t)i * H + j]; ss += v * v; }
  float norm    = sqrtf(ss);
  float desired = fminf(norm, 0.8f);
  float scale   = desired / (1e-7f + norm);
  _Float16* dst = Wt + (size_t)(g * H + j) * Ktot;
  for (int i = 0; i < H; ++i) dst[i] = (_Float16)(Whg[(size_t)i * H + j] * scale);
  const float* Wxg = Wx + (size_t)g * In * H;
  for (int i = 0; i < In; ++i) dst[H + i] = (_Float16)Wxg[(size_t)i * H + j];
}

// ---------------- generic WMMA GEMM: C(M,N) = epi(A(M,K)h @ Wt(N,K)h^T + bias) ----------------
// 256 threads (8 waves), 64x64 tile, double-buffered LDS staging (VGPR-pipelined global loads)
// Ping-pong buffers are one flat __shared__ array selected by integer offset (keeps ds_ addrspace).
// EPI: 0=f32, 1=f16, 2=tanh->f32, 3=exp + scatter (t,b,n)->(b,t,n)
#define GB_TILE (64 * SA_LD)
template <int EPI>
__global__ void __launch_bounds__(256)
gemm64(const _Float16* __restrict__ A, const _Float16* __restrict__ Wt,
       const float* __restrict__ bias, void* __restrict__ outp,
       int M, int N, int K) {
  __shared__ __align__(16) _Float16 sA[2 * GB_TILE];
  __shared__ __align__(16) _Float16 sB[2 * GB_TILE];
  int n0  = blockIdx.x * 64;
  int m0  = blockIdx.y * 64;
  int tid = threadIdx.x;
  int w   = tid >> 5;
  int mt  = w >> 1;
  int ntb = (w & 1) * 2;
  int row = tid >> 2;              // 0..63
  int seg = (tid & 3) * 8;         // 0,8,16,24 halves
  v8f acc[2] = {};
  float4 ra = *(const float4*)(A  + (size_t)(m0 + row) * K + seg);
  float4 rw = *(const float4*)(Wt + (size_t)(n0 + row) * K + seg);
  int p = 0;
  for (int kc = 0; kc < K; kc += 32) {
    *(float4*)(sA + p * GB_TILE + row * SA_LD + seg) = ra;
    *(float4*)(sB + p * GB_TILE + row * SA_LD + seg) = rw;
    if (kc + 32 < K) {                       // prefetch next chunk while this one computes
      ra = *(const float4*)(A  + (size_t)(m0 + row) * K + kc + 32 + seg);
      rw = *(const float4*)(Wt + (size_t)(n0 + row) * K + kc + 32 + seg);
    }
    __syncthreads();
    v16h a = load_frag_a(sA + p * GB_TILE + (mt * 16) * SA_LD, SA_LD);
#pragma unroll
    for (int q = 0; q < 2; ++q) {
      v16h bf = load_frag_b(sB + p * GB_TILE + ((ntb + q) * 16) * SA_LD, SA_LD);
      acc[q] = __builtin_amdgcn_wmma_f32_16x16x32_f16(false, a, false, bf,
                                                      (short)0, acc[q], false, false);
    }
    p ^= 1;   // alternating buffers: one barrier per iteration is sufficient
  }
  int lane  = tid & 31;
  int col   = lane & 15;
  int rhalf = (lane >> 4) * 8;
  for (int q = 0; q < 2; ++q) {
    int nG   = n0 + (ntb + q) * 16 + col;
    float bv = bias ? bias[nG] : 0.f;
#pragma unroll
    for (int vr = 0; vr < 8; ++vr) {
      int mG  = m0 + mt * 16 + rhalf + vr;
      float c = acc[q][vr] + bv;
      if (EPI == 0)      ((float*)outp)[(size_t)mG * N + nG] = c;
      else if (EPI == 1) ((_Float16*)outp)[(size_t)mG * N + nG] = (_Float16)c;
      else if (EPI == 2) ((float*)outp)[(size_t)mG * N + nG] = tanhf(c);
      else {
        int tt = mG / Bz, bb = mG % Bz;
        ((float*)outp)[((size_t)bb * Tz + tt) * N + nG] = __expf(c);
      }
    }
  }
}

// ---------------- scan helpers ----------------
// stage one 128x32 f16 activation tile into LDS via async DMA (2 x b128 per thread)
__device__ __forceinline__ void stage_A_async(const _Float16* Ah, int lda, int kc,
                                              _Float16* sAdst) {
  int tid  = threadIdx.x;
  int row  = tid >> 1;            // 0..127
  int part = (tid & 1) * 16;      // halves
  const _Float16* g = Ah + (size_t)row * lda + kc + part;
  _Float16* l = sAdst + row * SA_LD + part;
  async_copy_b128(l,     g);
  async_copy_b128(l + 8, g + 8);
}

// preload a 32-row weight panel (rows rowBase..rowBase+31 of Wt, each Ktot halves) into LDS
__device__ void preload_panel_async(const _Float16* Wt, int Ktot, int rowBase, _Float16* sW) {
  int nf4   = Ktot / 8;
  int total = 32 * nf4;           // multiple of 256 for all scans (Ktot in {512,576,768})
  for (int i = threadIdx.x; i < total; i += 256)
    async_copy_b128(sW + (i / nf4) * PANEL_LD + (i % nf4) * 8,
                    Wt + (size_t)(rowBase + i / nf4) * Ktot + (i % nf4) * 8);
}

// OUT(128, n0..n0+32) = Ah(128,Ktot) @ panel^T ; LDS-resident weights, async-double-buffered A.
// sAbase = flat 2*SA_TILE shared array; ping-pong by integer offset (no pointer selects).
__device__ void scan_gemm_panel(const _Float16* __restrict__ Ah, int Ktot,
                                const _Float16* sW,
                                float* __restrict__ OUT, int ldo, int n0,
                                _Float16* sAbase) {
  int tid = threadIdx.x;
  int w   = tid >> 5;             // wave w owns M rows [16w,16w+16)
  v8f acc[2] = {};
  stage_A_async(Ah, Ktot, 0, sAbase);
  int p = 0;
  int kc = 0;
  for (; kc + 32 < Ktot; kc += 32) {       // steady state: branch-free stage/compute
    stage_A_async(Ah, Ktot, kc + 32, sAbase + (p ^ 1) * SA_TILE);
    async_wait2();                         // the 2 older ops (current tile) are complete
    __syncthreads();
    v16h a = load_frag_a(sAbase + p * SA_TILE + (w * 16) * SA_LD, SA_LD);
#pragma unroll
    for (int nt = 0; nt < 2; ++nt) {
      v16h bf = load_frag_b(sW + (nt * 16) * PANEL_LD + kc, PANEL_LD);
      acc[nt] = __builtin_amdgcn_wmma_f32_16x16x32_f16(false, a, false, bf,
                                                       (short)0, acc[nt], false, false);
    }
    p ^= 1;
  }
  { // peeled last chunk
    async_wait0();
    __syncthreads();
    v16h a = load_frag_a(sAbase + p * SA_TILE + (w * 16) * SA_LD, SA_LD);
#pragma unroll
    for (int nt = 0; nt < 2; ++nt) {
      v16h bf = load_frag_b(sW + (nt * 16) * PANEL_LD + kc, PANEL_LD);
      acc[nt] = __builtin_amdgcn_wmma_f32_16x16x32_f16(false, a, false, bf,
                                                       (short)0, acc[nt], false, false);
    }
  }
  int lane  = tid & 31;
  int col   = lane & 15;
  int rhalf = (lane >> 4) * 8;
#pragma unroll
  for (int nt = 0; nt < 2; ++nt)
#pragma unroll
    for (int vr = 0; vr < 8; ++vr)
      OUT[(size_t)(w * 16 + rhalf + vr) * ldo + n0 + nt * 16 + col] = acc[nt][vr];
}

// ---------------- persistent cooperative GRU scan ----------------
// grid = 2H/32 WGs. Per WG, its 32-column weight panels (z|r block and c block) live in LDS
// for the whole T=400 scan. Per step: GEMM[z|r] -> sync -> sigmoid/r*h -> sync -> GEMM[c]
// -> sync -> h update -> sync. A1=[h|x_t], A2=[r*h|x_t]; Wt rows = output columns.
__global__ void __launch_bounds__(256)
gru_scan(const _Float16* __restrict__ xsrc, int In, int H, int fwd,
         const _Float16* __restrict__ Wt, const float* __restrict__ bias3,
         const float* __restrict__ h0,
         _Float16* __restrict__ A1, _Float16* __restrict__ A2,
         float* __restrict__ h32, float* __restrict__ zr, float* __restrict__ cpre,
         _Float16* __restrict__ outH, int outLd, int outOff, int outMode,
         unsigned* counter) {
  __shared__ __align__(16) _Float16 sW1[32 * PANEL_LD];   // [z|r] panel
  __shared__ __align__(16) _Float16 sW2[32 * PANEL_LD];   // [c] panel
  __shared__ __align__(16) _Float16 sAbuf[2 * SA_TILE];   // flat ping-pong A tiles
  const int nwg     = gridDim.x;      // 2H/32
  const int wg      = blockIdx.x;
  const int Ktot    = H + In;
  const int gstride = nwg * 256;
  const int gtid    = wg * 256 + threadIdx.x;
  const float* bz = bias3;
  const float* br = bias3 + H;
  const float* bc = bias3 + 2 * H;

  // ---- one-time LDS-resident weight panels ----
  preload_panel_async(Wt, Ktot, wg * 32, sW1);
  if (wg < H / 32) preload_panel_async(Wt, Ktot, 2 * H + wg * 32, sW2);
  async_wait0();
  __syncthreads();

  // ---- init: h = h0 (or 0), x part of A1/A2 = x_{t0} ----
  for (int idx = gtid; idx < Bz * H; idx += gstride) {
    int b = idx / H, j = idx % H;
    float hv = h0 ? h0[idx] : 0.f;
    h32[idx] = hv;
    A1[(size_t)b * Ktot + j] = (_Float16)hv;
  }
  int t0 = fwd ? 0 : (Tz - 1);
  for (int idx = gtid; idx < Bz * In; idx += gstride) {
    int b = idx / In, i = idx % In;
    _Float16 xv = xsrc[((size_t)t0 * Bz + b) * In + i];
    A1[(size_t)b * Ktot + H + i] = xv;
    A2[(size_t)b * Ktot + H + i] = xv;
  }
  grid_sync(counter, nwg);

  for (int s = 0; s < Tz; ++s) {
    int t = fwd ? s : (Tz - 1 - s);
    // GEMM1: zr_pre(B, 2H) = A1 @ [Wz|Wr]
    scan_gemm_panel(A1, Ktot, sW1, zr, 2 * H, wg * 32, sAbuf);
    grid_sync(counter, nwg);
    // EW1: z, r; stash z; A2 <- r*h
    for (int idx = gtid; idx < Bz * H; idx += gstride) {
      int b = idx / H, j = idx % H;
      float zp = zr[(size_t)b * 2 * H + j] + bz[j];
      float rp = zr[(size_t)b * 2 * H + H + j] + br[j];
      float z = 1.f / (1.f + __expf(-zp));
      float r = 1.f / (1.f + __expf(-rp));
      zr[(size_t)b * 2 * H + j] = z;
      A2[(size_t)b * Ktot + j] = (_Float16)(r * h32[idx]);
    }
    grid_sync(counter, nwg);
    // GEMM2: c_pre(B, H) = A2 @ Wc
    if (wg < H / 32)
      scan_gemm_panel(A2, Ktot, sW2, cpre, H, wg * 32, sAbuf);
    grid_sync(counter, nwg);
    // EW2: h_new = (1-z)h + z*tanh(c_pre + bc); stage next x
    for (int idx = gtid; idx < Bz * H; idx += gstride) {
      int b = idx / H, j = idx % H;
      float z  = zr[(size_t)b * 2 * H + j];
      float ht = tanhf(cpre[idx] + bc[j]);
      float hn = (1.f - z) * h32[idx] + z * ht;
      h32[idx] = hn;
      A1[(size_t)b * Ktot + j] = (_Float16)hn;
      if (outMode == 1)
        outH[((size_t)t * Bz + b) * outLd + outOff + j] = (_Float16)hn;
      else if (outMode == 2 && s == Tz - 1)
        outH[(size_t)b * outLd + outOff + j] = (_Float16)hn;
    }
    if (s + 1 < Tz) {
      int tn = fwd ? (s + 1) : (Tz - 2 - s);
      for (int idx = gtid; idx < Bz * In; idx += gstride) {
        int b = idx / In, i = idx % In;
        _Float16 xv = xsrc[((size_t)tn * Bz + b) * In + i];
        A1[(size_t)b * Ktot + H + i] = xv;
        A2[(size_t)b * Ktot + H + i] = xv;
      }
    }
    grid_sync(counter, nwg);
  }
}

// ---------------- variational sampling elementwise ----------------
__global__ void zic_ew(const float* __restrict__ icm, const float* __restrict__ icl,
                       const float* __restrict__ eps, _Float16* __restrict__ zich, int n) {
  int i = blockIdx.x * blockDim.x + threadIdx.x;
  if (i >= n) return;
  float lv = fminf(fmaxf(icl[i], -10.f), 10.f);
  zich[i] = (_Float16)(icm[i] + eps[i] * __expf(0.5f * lv));
}

__global__ void us_ew(const float* __restrict__ um, const float* __restrict__ ul,
                      const float* __restrict__ epsu, _Float16* __restrict__ ush) {
  int idx = blockIdx.x * blockDim.x + threadIdx.x;
  const int total = Tz * Bz * Uz;
  if (idx >= total) return;
  int u  = idx % Uz;
  int tb = idx / Uz;
  int b  = tb % Bz;
  int t  = tb / Bz;
  float lv = fminf(fmaxf(ul[idx], -10.f), 10.f);
  ush[idx] = (_Float16)(um[idx] + epsu[((size_t)b * Tz + t) * Uz + u] * __expf(0.5f * lv));
}

// ---------------- host-side orchestration ----------------
extern "C" void kernel_launch(void* const* d_in, const int* in_sizes, int n_in,
                              void* d_out, int out_size, void* d_ws, size_t ws_size,
                              hipStream_t stream) {
  (void)in_sizes; (void)n_in; (void)out_size; (void)ws_size;
  const float* x        = (const float*)d_in[0];
  const float* eps_ic   = (const float*)d_in[1];
  const float* eps_u    = (const float*)d_in[2];
  const float* encf_Wx  = (const float*)d_in[3];
  const float* encf_Wh  = (const float*)d_in[4];
  const float* encf_b   = (const float*)d_in[5];
  const float* encb_Wx  = (const float*)d_in[6];
  const float* encb_Wh  = (const float*)d_in[7];
  const float* encb_b   = (const float*)d_in[8];
  const float* conf_Wx  = (const float*)d_in[9];
  const float* conf_Wh  = (const float*)d_in[10];
  const float* conf_b   = (const float*)d_in[11];
  const float* conb_Wx  = (const float*)d_in[12];
  const float* conb_Wh  = (const float*)d_in[13];
  const float* conb_b   = (const float*)d_in[14];
  const float* gen_Wx   = (const float*)d_in[15];
  const float* gen_Wh   = (const float*)d_in[16];
  const float* gen_b    = (const float*)d_in[17];
  const float* icm_W    = (const float*)d_in[18];
  const float* icm_b    = (const float*)d_in[19];
  const float* icl_W    = (const float*)d_in[20];
  const float* icl_b    = (const float*)d_in[21];
  const float* um_W     = (const float*)d_in[22];
  const float* um_b     = (const float*)d_in[23];
  const float* ul_W     = (const float*)d_in[24];
  const float* ul_b     = (const float*)d_in[25];
  const float* init_W   = (const float*)d_in[26];
  const float* init_b   = (const float*)d_in[27];
  const float* fac_W    = (const float*)d_in[28];
  const float* fac_b    = (const float*)d_in[29];
  const float* obs_W    = (const float*)d_in[30];
  const float* obs_b    = (const float*)d_in[31];
  float* out = (float*)d_out;

  char* ws = (char*)d_ws;
  size_t off = 0;
  auto alloc = [&](size_t bytes) -> char* {
    char* p = ws + off;
    off += (bytes + 255) & ~(size_t)255;
    return p;
  };

  unsigned*  cnt    = (unsigned*)alloc(64 * sizeof(unsigned));
  _Float16*  x_h    = (_Float16*)alloc((size_t)Tz * Bz * Nz * 2);
  _Float16*  Wef    = (_Float16*)alloc((size_t)3 * EHz * (EHz + Nz) * 2);
  _Float16*  Web    = (_Float16*)alloc((size_t)3 * EHz * (EHz + Nz) * 2);
  _Float16*  Wcf    = (_Float16*)alloc((size_t)3 * CHz * (CHz + Nz) * 2);
  _Float16*  Wcb    = (_Float16*)alloc((size_t)3 * CHz * (CHz + Nz) * 2);
  _Float16*  Wg     = (_Float16*)alloc((size_t)3 * GHz * (GHz + Uz) * 2);
  _Float16*  icmW_t = (_Float16*)alloc((size_t)2 * EHz * ICz * 2);
  _Float16*  iclW_t = (_Float16*)alloc((size_t)2 * EHz * ICz * 2);
  _Float16*  umW_t  = (_Float16*)alloc((size_t)2 * CHz * Uz * 2);
  _Float16*  ulW_t  = (_Float16*)alloc((size_t)2 * CHz * Uz * 2);
  _Float16*  initW_t= (_Float16*)alloc((size_t)ICz * GHz * 2);
  _Float16*  facW_t = (_Float16*)alloc((size_t)GHz * FDz * 2);
  _Float16*  obsW_t = (_Float16*)alloc((size_t)FDz * Nz * 2);
  _Float16*  A1     = (_Float16*)alloc((size_t)Bz * (EHz + Nz) * 2);
  _Float16*  A2     = (_Float16*)alloc((size_t)Bz * (EHz + Nz) * 2);
  float*     h32    = (float*)alloc((size_t)Bz * GHz * 4);
  float*     zr     = (float*)alloc((size_t)Bz * 2 * GHz * 4);
  float*     cpre   = (float*)alloc((size_t)Bz * GHz * 4);
  _Float16*  cfin   = (_Float16*)alloc((size_t)Bz * 2 * EHz * 2);
  _Float16*  comb   = (_Float16*)alloc((size_t)Tz * Bz * 2 * CHz * 2);
  float*     icm    = (float*)alloc((size_t)Bz * ICz * 4);
  float*     icl    = (float*)alloc((size_t)Bz * ICz * 4);
  _Float16*  zic_h  = (_Float16*)alloc((size_t)Bz * ICz * 2);
  float*     g0     = (float*)alloc((size_t)Bz * GHz * 4);
  float*     um     = (float*)alloc((size_t)Tz * Bz * Uz * 4);
  float*     ul     = (float*)alloc((size_t)Tz * Bz * Uz * 4);
  _Float16*  us_h   = (_Float16*)alloc((size_t)Tz * Bz * Uz * 2);
  _Float16*  st_h   = (_Float16*)alloc((size_t)Tz * Bz * GHz * 2);
  _Float16*  fac_h  = (_Float16*)alloc((size_t)Tz * Bz * FDz * 2);

  // ---- prep ----
  zero_u32<<<1, 64, 0, stream>>>(cnt, 64);
  {
    int total = Tz * Bz * Nz;
    pack_x_kernel<<<(total + 255) / 256, 256, 0, stream>>>(x, x_h);
  }
  constrain_pack_kernel<<<(3 * EHz + 255) / 256, 256, 0, stream>>>(encf_Wh, encf_Wx, Wef, EHz, Nz);
  constrain_pack_kernel<<<(3 * EHz + 255) / 256, 256, 0, stream>>>(encb_Wh, encb_Wx, Web, EHz, Nz);
  constrain_pack_kernel<<<(3 * CHz + 255) / 256, 256, 0, stream>>>(conf_Wh, conf_Wx, Wcf, CHz, Nz);
  constrain_pack_kernel<<<(3 * CHz + 255) / 256, 256, 0, stream>>>(conb_Wh, conb_Wx, Wcb, CHz, Nz);
  constrain_pack_kernel<<<(3 * GHz + 255) / 256, 256, 0, stream>>>(gen_Wh, gen_Wx, Wg, GHz, Uz);
  f2h_t_kernel<<<(2 * EHz * ICz + 255) / 256, 256, 0, stream>>>(icm_W, icmW_t, 2 * EHz, ICz);
  f2h_t_kernel<<<(2 * EHz * ICz + 255) / 256, 256, 0, stream>>>(icl_W, iclW_t, 2 * EHz, ICz);
  f2h_t_kernel<<<(2 * CHz * Uz + 255) / 256, 256, 0, stream>>>(um_W, umW_t, 2 * CHz, Uz);
  f2h_t_kernel<<<(2 * CHz * Uz + 255) / 256, 256, 0, stream>>>(ul_W, ulW_t, 2 * CHz, Uz);
  f2h_t_kernel<<<(ICz * GHz + 255) / 256, 256, 0, stream>>>(init_W, initW_t, ICz, GHz);
  f2h_t_kernel<<<(GHz * FDz + 255) / 256, 256, 0, stream>>>(fac_W, facW_t, GHz, FDz);
  f2h_t_kernel<<<(FDz * Nz + 255) / 256, 256, 0, stream>>>(obs_W, obsW_t, FDz, Nz);

  // ---- variational (IC) encoder BiGRU: keep only final states ----
  gru_scan<<<dim3(2 * EHz / 32), dim3(256), 0, stream>>>(
      x_h, Nz, EHz, 1, Wef, encf_b, nullptr, A1, A2, h32, zr, cpre,
      cfin, 2 * EHz, 0, 2, cnt + 0);
  gru_scan<<<dim3(2 * EHz / 32), dim3(256), 0, stream>>>(
      x_h, Nz, EHz, 0, Web, encb_b, nullptr, A1, A2, h32, zr, cpre,
      cfin, 2 * EHz, EHz, 2, cnt + 1);

  // ---- q(z_ic) heads, sample, g0 = tanh(fc_init(z)) ----
  gemm64<0><<<dim3(ICz / 64, Bz / 64), dim3(256), 0, stream>>>(cfin, icmW_t, icm_b, icm, Bz, ICz, 2 * EHz);
  gemm64<0><<<dim3(ICz / 64, Bz / 64), dim3(256), 0, stream>>>(cfin, iclW_t, icl_b, icl, Bz, ICz, 2 * EHz);
  zic_ew<<<(Bz * ICz + 255) / 256, 256, 0, stream>>>(icm, icl, eps_ic, zic_h, Bz * ICz);
  gemm64<2><<<dim3(GHz / 64, Bz / 64), dim3(256), 0, stream>>>(zic_h, initW_t, init_b, g0, Bz, GHz, ICz);

  // ---- controller BiGRU: store every state into comb (T,B,2CH) ----
  gru_scan<<<dim3(2 * CHz / 32), dim3(256), 0, stream>>>(
      x_h, Nz, CHz, 1, Wcf, conf_b, nullptr, A1, A2, h32, zr, cpre,
      comb, 2 * CHz, 0, 1, cnt + 2);
  gru_scan<<<dim3(2 * CHz / 32), dim3(256), 0, stream>>>(
      x_h, Nz, CHz, 0, Wcb, conb_b, nullptr, A1, A2, h32, zr, cpre,
      comb, 2 * CHz, CHz, 1, cnt + 3);

  // ---- q(u_t) heads + sampling ----
  gemm64<0><<<dim3(Uz / 64, Tz * Bz / 64), dim3(256), 0, stream>>>(comb, umW_t, um_b, um, Tz * Bz, Uz, 2 * CHz);
  gemm64<0><<<dim3(Uz / 64, Tz * Bz / 64), dim3(256), 0, stream>>>(comb, ulW_t, ul_b, ul, Tz * Bz, Uz, 2 * CHz);
  us_ew<<<(Tz * Bz * Uz + 255) / 256, 256, 0, stream>>>(um, ul, eps_u, us_h);

  // ---- generator GRU from g0, inputs u_t ----
  gru_scan<<<dim3(2 * GHz / 32), dim3(256), 0, stream>>>(
      us_h, Uz, GHz, 1, Wg, gen_b, g0, A1, A2, h32, zr, cpre,
      st_h, GHz, 0, 1, cnt + 4);

  // ---- factors, then rates = exp(obs(factors)) scattered to (B,T,N) ----
  gemm64<1><<<dim3(FDz / 64, Tz * Bz / 64), dim3(256), 0, stream>>>(st_h, facW_t, fac_b, fac_h, Tz * Bz, FDz, GHz);
  gemm64<3><<<dim3(Nz / 64, Tz * Bz / 64), dim3(256), 0, stream>>>(fac_h, obsW_t, obs_b, out, Tz * Bz, Nz, FDz);
}
